// RecurrentNet_82815559402249
// MI455X (gfx1250) — compile-verified, matrix operand order checked by reading, not run on hardware
//
#include <hip/hip_runtime.h>
#include <hip/hip_bf16.h>

// ---------------------------------------------------------------------------
// RecurrentNet for MI455X (gfx1250, wave32, WMMA).
//   A1 = x @ (Wfc@(W1x_lo+W1x_hi)) + bfc@(...)   (bulk parallel, bf16 WMMA)
//   A2 = x @ (Wfc@(W2x_lo+W2x_hi)) + bfc@(...)
//   scan: h' = tanh(A1[t] + h@W1h + sigmoid(c@W1g)*h)
//         c' = sigmoid(A2[t] + c@W2h) * tanh(h@W2g)
// Persistent 8-block grid for the scan; 64 waves == 2 streams x 32 N-tiles,
// each wave: 32x v_wmma_f32_16x16x32_bf16 per timestep.
//  - Recurrent weights live in LDS (256KB/block + 33KB state staging fits the
//    320KB WGP LDS of CDNA5); B-fragments come from ds_load_b128.
//  - h/c state and the one-time weight preload use GLOBAL_LOAD_ASYNC_TO_LDS
//    (ASYNCcnt) + s_wait_asynccnt, the CDNA5 async copy path.
//  - Grid barrier: agent-scope release/acquire atomic + s_sleep.
// ---------------------------------------------------------------------------

typedef __attribute__((ext_vector_type(8)))  __bf16 v8bf;
typedef __attribute__((ext_vector_type(16))) __bf16 v16bf;
typedef __attribute__((ext_vector_type(8)))  float  v8f;

#define BS   16
#define SEQ  1024
#define IN_  512
#define H_   512
#define KT   16                 // K tiles of 32 over 512
#define NTL  32                 // N tiles of 16 over 512
#define MATE (512*512)          // elements per (packed) 512x512 matrix
#define GBLK 8                  // persistent blocks in scan kernel

// LDS layout for k_scan (elements of bf16):
//   hs[16][520], cs[16][520], then 8 waves x 16384 (two 16KB weight slices)
#define LDS_STATE_EL   (2 * 16 * 520)
#define LDS_WSLICE_EL  (2 * KT * 512)          // 16384 bf16 = 32KB per wave
#define SCAN_LDS_BYTES ((LDS_STATE_EL + 8 * LDS_WSLICE_EL) * 2)

__device__ __forceinline__ float sigmoidf_(float x) {
    return 1.0f / (1.0f + __expf(-x));
}
__device__ __forceinline__ float tanhf_(float x) {
    x = fminf(fmaxf(x, -15.0f), 15.0f);
    float e = __expf(2.0f * x);
    return (e - 1.0f) / (e + 1.0f);
}

__device__ __forceinline__ v16bf cat16(v8bf lo, v8bf hi) {
    return __builtin_shufflevector(lo, hi, 0,1,2,3,4,5,6,7,8,9,10,11,12,13,14,15);
}

// --- CDNA5 async global->LDS copy (ASYNCcnt), GVS addressing ----------------
// LDS byte address in a VGPR; global address is the SGPR base plus a 32-bit VGPR offset
__device__ __forceinline__ void async_ld_b128(unsigned lds_off, const void* base,
                                              unsigned byte_off) {
    asm volatile("global_load_async_to_lds_b128 %0, %1, %2"
                 :: "v"(lds_off), "v"(byte_off), "s"(base) : "memory");
}
__device__ __forceinline__ void wait_async0() {
    asm volatile("s_wait_asynccnt 0" ::: "memory");
}

// A-fragment (16x32 bf16, M x K) built from a row-major bf16 row pointer.
// Lane layout (ISA 7.12.2): row = lane&15; elems 0..7 -> K = kb + 8*(lane>>4) + e,
// elems 8..15 -> K = kb + 16 + 8*(lane>>4) + (e-8).  Two 16B loads.
__device__ __forceinline__ v16bf afrag_bf16(const __bf16* row, int kb, int half) {
    v8bf lo = *reinterpret_cast<const v8bf*>(row + kb + half * 8);
    v8bf hi = *reinterpret_cast<const v8bf*>(row + kb + 16 + half * 8);
    return cat16(lo, hi);
}

// Same A-fragment but source is fp32 (converted to bf16 on the fly).
__device__ __forceinline__ v16bf afrag_f32(const float* row, int kb, int half) {
    const float* p0 = row + kb + half * 8;
    const float* p1 = row + kb + 16 + half * 8;
    float4 a0 = *reinterpret_cast<const float4*>(p0);
    float4 a1 = *reinterpret_cast<const float4*>(p0 + 4);
    float4 b0 = *reinterpret_cast<const float4*>(p1);
    float4 b1 = *reinterpret_cast<const float4*>(p1 + 4);
    v16bf r;
    r[0]=(__bf16)a0.x; r[1]=(__bf16)a0.y; r[2]=(__bf16)a0.z; r[3]=(__bf16)a0.w;
    r[4]=(__bf16)a1.x; r[5]=(__bf16)a1.y; r[6]=(__bf16)a1.z; r[7]=(__bf16)a1.w;
    r[8]=(__bf16)b0.x; r[9]=(__bf16)b0.y; r[10]=(__bf16)b0.z; r[11]=(__bf16)b0.w;
    r[12]=(__bf16)b1.x; r[13]=(__bf16)b1.y; r[14]=(__bf16)b1.z; r[15]=(__bf16)b1.w;
    return r;
}

// B-fragment from a weight matrix pre-swizzled into fragment order:
// 512 bf16 per (nt,kt) tile, 16 contiguous per lane -> two 16B loads.
__device__ __forceinline__ v16bf bfrag(const __bf16* mat, int nt, int kt, int lane) {
    const v8bf* p = reinterpret_cast<const v8bf*>(
        mat + (size_t)(((nt * KT + kt) * 32 + lane)) * 16);
    return cat16(p[0], p[1]);
}

// B-fragment from this wave's LDS weight slice: mat in {0,1}
__device__ __forceinline__ v16bf bfrag_lds(const __bf16* wbase, int mat, int kt,
                                           int lane) {
    const v8bf* p = reinterpret_cast<const v8bf*>(
        wbase + ((mat * KT + kt) * 512 + lane * 16));
    return cat16(p[0], p[1]);
}

__device__ __forceinline__ v8f wmma_bf16(v16bf a, v16bf b, v8f c) {
    return __builtin_amdgcn_wmma_f32_16x16x32_bf16(
        /*neg_a=*/false, a, /*neg_b=*/false, b,
        /*c_mod=*/(short)0, c, /*reuse_a=*/false, /*reuse_b=*/false);
}

// --- weight combine: Wc[g] = Wfc @ (Wx_lo + Wx_hi), fp32 (one-time, tiny) ---
__global__ __launch_bounds__(256) void k_combine_w(
    const float* __restrict__ Wfc, const float* __restrict__ W1x,
    const float* __restrict__ W2x, float* __restrict__ wc) {
    unsigned t = blockIdx.x * 256u + threadIdx.x;   // 2*512*512 threads
    unsigned g = t >> 18, rem = t & 0x3FFFFu;
    unsigned i = rem >> 9, n = rem & 511u;
    const float* Wx = g ? W2x : W1x;
    float acc = 0.0f;
    for (int k = 0; k < 512; ++k)
        acc = fmaf(Wfc[i * 512 + k], Wx[k * 512 + n] + Wx[(k + 512) * 512 + n], acc);
    wc[(size_t)g * MATE + i * 512 + n] = acc;
}

__global__ __launch_bounds__(256) void k_combine_b(
    const float* __restrict__ bfc, const float* __restrict__ W1x,
    const float* __restrict__ W2x, float* __restrict__ bc) {
    unsigned t = blockIdx.x * 256u + threadIdx.x;
    if (t >= 1024u) return;
    unsigned g = t >> 9, n = t & 511u;
    const float* Wx = g ? W2x : W1x;
    float acc = 0.0f;
    for (int k = 0; k < 512; ++k)
        acc = fmaf(bfc[k], Wx[k * 512 + n] + Wx[(k + 512) * 512 + n], acc);
    bc[t] = acc;
}

// --- pack 6 weight matrices (Wc1, Wc2, W1h, W1g, W2h, W2g) into bf16
//     WMMA B-fragment order ------------------------------------------------
__global__ __launch_bounds__(256) void k_pack(
    const float* __restrict__ wc, const float* __restrict__ W1h,
    const float* __restrict__ W1g, const float* __restrict__ W2h,
    const float* __restrict__ W2g, __bf16* __restrict__ packed) {
    unsigned t = blockIdx.x * 256u + threadIdx.x;   // 6*16*32*32 threads
    unsigned lane = t & 31u;
    unsigned w = t >> 5;
    unsigned nt = w & 31u;
    unsigned w2 = w >> 5;
    unsigned kt = w2 & 15u;
    unsigned g = w2 >> 4;                            // 0..5
    const float* src;
    switch (g) {
        case 0: src = wc;          break;
        case 1: src = wc + MATE;   break;
        case 2: src = W1h;         break;
        case 3: src = W1g;         break;
        case 4: src = W2h;         break;
        default: src = W2g;        break;
    }
    unsigned half = lane >> 4, n = nt * 16 + (lane & 15u), kb = kt * 32;
    __bf16* dst = packed + (size_t)g * MATE +
                  (size_t)((nt * KT + kt) * 32 + lane) * 16;
    for (int e = 0; e < 16; ++e) {
        unsigned K = kb + half * 8 + ((e < 8) ? (unsigned)e : (unsigned)(e + 8));
        dst[e] = (__bf16)src[(size_t)K * 512 + n];
    }
}

// --- zero h0/c0 ping-pong state + barrier counter ---------------------------
__global__ void k_init(__bf16* hc, unsigned* cnt) {
    for (int i = threadIdx.x; i < 4 * BS * H_; i += 256) hc[i] = (__bf16)0.0f;
    if (threadIdx.x == 0) *cnt = 0u;
}

// --- bulk input GEMMs: A1 = x@Wc1 + b1, A2 = x@Wc2 + b2 ---------------------
__global__ __launch_bounds__(256) void k_gemm_in(
    const float* __restrict__ x, const __bf16* __restrict__ packed,
    const float* __restrict__ bc, float* __restrict__ A1, float* __restrict__ A2) {
    unsigned wid = blockIdx.x * 8u + (threadIdx.x >> 5);   // 65536 waves total
    unsigned lane = threadIdx.x & 31u;
    unsigned which = wid >> 15;                            // 0 -> A1, 1 -> A2
    unsigned rem = wid & 32767u;
    unsigned mt = rem >> 5, nt = rem & 31u;
    const __bf16* wp = packed + (size_t)which * MATE;
    const float* bias = bc + which * 512;
    float* Aout = which ? A2 : A1;
    unsigned m = lane & 15u, half = lane >> 4;
    const float* xrow = x + (size_t)(mt * 16 + m) * IN_;
    v8f acc = {};
    for (int kt = 0; kt < KT; ++kt) {
        v16bf a = afrag_f32(xrow, kt * 32, (int)half);
        v16bf b = bfrag(wp, (int)nt, kt, (int)lane);
        acc = wmma_bf16(a, b, acc);
    }
    unsigned n = nt * 16 + m;
    float bv = bias[n];
    for (int v = 0; v < 8; ++v) {                          // C layout: M = v + 8*half
        unsigned row = mt * 16 + v + half * 8;
        Aout[(size_t)row * H_ + n] = acc[v] + bv;
    }
}

// --- persistent recurrent scan ---------------------------------------------
// NOTE: uses 295KB dynamic LDS (fits CDNA5's 320KB/WGP). A real host driver
// would set hipFuncAttributeMaxDynamicSharedMemorySize once before capture.
__global__ __launch_bounds__(256) void k_scan(
    const float* __restrict__ A1, const float* __restrict__ A2,
    const __bf16* __restrict__ packed, __bf16* __restrict__ hc,
    float* __restrict__ out, unsigned* __restrict__ cnt) {
    extern __shared__ __bf16 smem[];
    __bf16 (*hs)[520] = reinterpret_cast<__bf16(*)[520]>(smem);
    __bf16 (*cs)[520] = reinterpret_cast<__bf16(*)[520]>(smem + 16 * 520);
    __bf16* wlds = smem + LDS_STATE_EL;

    const int tid  = threadIdx.x;
    const int lane = tid & 31;
    const int wave = tid >> 5;
    const int wid  = blockIdx.x * 8 + wave;         // 0..63
    const int strm = wid >> 5;                      // 0: h-stream, 1: c-stream
    const int nt   = wid & 31;
    // strm0: acc1 = h@W1h (g2), acc2 = c@W1g (g3)
    // strm1: acc1 = c@W2h (g4), acc2 = h@W2g (g5)
    const __bf16* wA = packed + (size_t)(2 + strm * 2) * MATE;
    const __bf16* wB = packed + (size_t)(3 + strm * 2) * MATE;
    __bf16* wl = wlds + (size_t)wave * LDS_WSLICE_EL;

    const int m = lane & 15, half = lane >> 4;
    const int n = nt * 16 + m;
    const int srow = tid >> 4;                      // staging: 64B per thread
    const int scol = (tid & 15) * 32;

    // One-time async preload of this wave's two weight slices into LDS
    // (completion folded into the first step's s_wait_asynccnt 0).
    for (int mat = 0; mat < 2; ++mat) {
        const void* src = (mat == 0) ? (const void*)wA : (const void*)wB;
        for (int kt = 0; kt < KT; ++kt) {
            unsigned srcOff = (unsigned)((((nt * KT + kt) * 32 + lane) * 16) * 2);
            unsigned ldsOff = (unsigned)(uintptr_t)
                (wl + (mat * KT + kt) * 512 + lane * 16);
            async_ld_b128(ldsOff, src, srcOff);
            async_ld_b128(ldsOff + 16, src, srcOff + 16);
        }
    }

    for (int t = 0; t < SEQ; ++t) {
        const __bf16* hcur = hc + (size_t)(t & 1) * (BS * H_);
        const __bf16* ccur = hc + (size_t)(2 + (t & 1)) * (BS * H_);
        __bf16* hnxt = hc + (size_t)((t + 1) & 1) * (BS * H_);
        __bf16* cnxt = hc + (size_t)(2 + ((t + 1) & 1)) * (BS * H_);

        // async-stage h,c (16x512 bf16 each) into LDS, then retire ASYNCcnt
        {
            unsigned gOff = (unsigned)((srow * H_ + scol) * 2);
            for (int j = 0; j < 4; ++j) {
                async_ld_b128((unsigned)(uintptr_t)&hs[srow][scol + j * 8],
                              hcur, gOff + j * 16);
                async_ld_b128((unsigned)(uintptr_t)&cs[srow][scol + j * 8],
                              ccur, gOff + j * 16);
            }
        }
        wait_async0();
        __syncthreads();

        v8f acc1 = {}, acc2 = {};
        const __bf16* rA = (strm == 0) ? &hs[m][0] : &cs[m][0];
        const __bf16* rB = (strm == 0) ? &cs[m][0] : &hs[m][0];
        for (int kt = 0; kt < KT; ++kt) {
            v16bf aA = afrag_bf16(rA, kt * 32, half);
            acc1 = wmma_bf16(aA, bfrag_lds(wl, 0, kt, lane), acc1);
            v16bf aB = afrag_bf16(rB, kt * 32, half);
            acc2 = wmma_bf16(aB, bfrag_lds(wl, 1, kt, lane), acc2);
        }

        // prefetch next timestep's input-activation rows (global_prefetch)
        if (t + 1 < SEQ) {
            const float* pf = (strm == 0 ? A1 : A2) +
                              ((size_t)(half * 8) * SEQ + (t + 1)) * H_ + n;
            __builtin_prefetch(pf, 0, 1);
        }

        if (strm == 0) {
            for (int v = 0; v < 8; ++v) {
                int b = v + half * 8;
                float a1   = A1[((size_t)b * SEQ + t) * H_ + n];
                float hold = (float)hs[b][n];
                float hn   = tanhf_(a1 + acc1[v] + sigmoidf_(acc2[v]) * hold);
                out[((size_t)b * SEQ + t) * H_ + n] = hn;          // hseq[b][t][n]
                hnxt[b * H_ + n] = (__bf16)hn;
                if (t == SEQ - 1)
                    out[(size_t)BS * SEQ * H_ + b * H_ + n] = hn;  // h_fin
            }
        } else {
            for (int v = 0; v < 8; ++v) {
                int b = v + half * 8;
                float a2v = A2[((size_t)b * SEQ + t) * H_ + n];
                float cn  = sigmoidf_(a2v + acc1[v]) * tanhf_(acc2[v]);
                cnxt[b * H_ + n] = (__bf16)cn;
                if (t == SEQ - 1)
                    out[(size_t)BS * SEQ * H_ + BS * H_ + b * H_ + n] = cn; // c_fin
            }
        }

        // grid-wide step barrier (release stores, acquire counter)
        __threadfence();
        __syncthreads();
        if (tid == 0) {
            __hip_atomic_fetch_add(cnt, 1u, __ATOMIC_ACQ_REL, __HIP_MEMORY_SCOPE_AGENT);
            unsigned tgt = (unsigned)GBLK * (unsigned)(t + 1);
            while (__hip_atomic_load(cnt, __ATOMIC_ACQUIRE, __HIP_MEMORY_SCOPE_AGENT) < tgt)
                __builtin_amdgcn_s_sleep(2);
        }
        __syncthreads();
    }
}

extern "C" void kernel_launch(void* const* d_in, const int* in_sizes, int n_in,
                              void* d_out, int out_size, void* d_ws, size_t ws_size,
                              hipStream_t stream) {
    (void)in_sizes; (void)n_in; (void)out_size; (void)ws_size;
    const float* x   = (const float*)d_in[0];
    const float* Wfc = (const float*)d_in[1];
    const float* bfc = (const float*)d_in[2];
    const float* W1x = (const float*)d_in[3];
    const float* W1h = (const float*)d_in[4];
    const float* W1g = (const float*)d_in[5];
    const float* W2x = (const float*)d_in[6];
    const float* W2h = (const float*)d_in[7];
    const float* W2g = (const float*)d_in[8];
    float* out = (float*)d_out;

    char* ws = (char*)d_ws;
    float*   wc     = (float*)ws;                                   // 2*MATE f32
    float*   bc     = (float*)(ws + (size_t)2 * MATE * 4);          // 1024 f32
    __bf16*  packed = (__bf16*)(ws + (size_t)2 * MATE * 4 + 4096);  // 6*MATE bf16
    float*   A1     = (float*)((char*)packed + (size_t)6 * MATE * 2);
    float*   A2     = A1 + (size_t)BS * SEQ * H_;
    __bf16*  hc     = (__bf16*)(A2 + (size_t)BS * SEQ * H_);        // 4*BS*H bf16
    unsigned* cnt   = (unsigned*)(hc + 4 * BS * H_);

    k_combine_w<<<2048, 256, 0, stream>>>(Wfc, W1x, W2x, wc);
    k_combine_b<<<4,    256, 0, stream>>>(bfc, W1x, W2x, bc);
    k_pack     <<<384,  256, 0, stream>>>(wc, W1h, W1g, W2h, W2g, packed);
    k_init     <<<1,    256, 0, stream>>>(hc, cnt);
    k_gemm_in  <<<8192, 256, 0, stream>>>(x, packed, bc, A1, A2);
    k_scan     <<<GBLK, 256, SCAN_LDS_BYTES, stream>>>(A1, A2, packed, hc, out, cnt);
}
